// equinet_fstar_4269197492354
// MI455X (gfx1250) — matrix operand order repaired and checked by reading, not compiled, not run on hardware
//
#include <hip/hip_runtime.h>

typedef __attribute__((ext_vector_type(2)))  float  v2f;
typedef __attribute__((ext_vector_type(8)))  float  v8f;
typedef __attribute__((ext_vector_type(8)))  __bf16 v8bf;
typedef __attribute__((ext_vector_type(16))) __bf16 v16bf;

#define NXX 80
#define NX2 6400
#define NBATCH 1600
#define NG 20          // 1600*80 / 6400
#define KP 96          // K padded to 3 x 32 for bf16 WMMA
#define SSTR 104       // LDS row stride (bf16 elems): 52-dword lane stride -> conflict-free b128
#define NBT (4 * NXX * KP)   // elems per Bt plane

// ---------------- bf16 split helpers (round-to-nearest-even) ----------------
__device__ __forceinline__ unsigned short f2bf_bits(float x) {
    unsigned u = __float_as_uint(x);
    unsigned r = u + 0x7FFFu + ((u >> 16) & 1u);
    return (unsigned short)(r >> 16);
}
__device__ __forceinline__ float bfbits2f(unsigned short h) {
    return __uint_as_float(((unsigned)h) << 16);
}
__device__ __forceinline__ void split_bits(float x, unsigned short& hb, unsigned short& lb) {
    hb = f2bf_bits(x);
    lb = f2bf_bits(x - bfbits2f(hb));
}
__device__ __forceinline__ __bf16 bc(unsigned short u) {
    return __builtin_bit_cast(__bf16, u);
}
// two 16B chunks (gap in elements) -> one 16-elem bf16 fragment
__device__ __forceinline__ v16bf ld2x8(const __bf16* p, int gap) {
    v8bf a = *(const v8bf*)p;
    v8bf b = *(const v8bf*)(p + gap);
    return __builtin_shufflevector(a, b, 0,1,2,3,4,5,6,7,8,9,10,11,12,13,14,15);
}

// -------- Kernel 1: build A_t = post_t (x) K2_t (f32) and split-transposed
//          Bt_t[j,k] = pre_t[k]*K1_t[k,j] as bf16 hi/lo planes, K padded to 96 --------
__global__ __launch_bounds__(256) void eqnet_precompute(
    const float* __restrict__ q0, const float* __restrict__ q1,
    const float* __restrict__ q2, const float* __restrict__ q3,   // post per term
    const float* __restrict__ k20, const float* __restrict__ k21,
    const float* __restrict__ k22, const float* __restrict__ k23, // K2 per term
    const float* __restrict__ p0, const float* __restrict__ p1,
    const float* __restrict__ p2, const float* __restrict__ p3,   // pre per term
    const float* __restrict__ k10, const float* __restrict__ k11,
    const float* __restrict__ k12, const float* __restrict__ k13, // K1 per term
    float* __restrict__ wsA, __bf16* __restrict__ wsBth, __bf16* __restrict__ wsBtl)
{
    int idx = blockIdx.x * 256 + threadIdx.x;
    if (idx < 4 * NX2) {
        int t = idx / NX2;
        int e = idx - t * NX2;
        int i = e / NXX;
        float v;
        switch (t) {
            case 0:  v = q0[i] * k20[e]; break;
            case 1:  v = q1[i] * k21[e]; break;
            case 2:  v = q2[i] * k22[e]; break;
            default: v = q3[i] * k23[e]; break;
        }
        wsA[idx] = v;
    } else if (idx < 4 * NX2 + NBT) {
        int r = idx - 4 * NX2;
        int term = r / (NXX * KP);
        int rr = r - term * (NXX * KP);
        int j = rr / KP, k = rr - j * KP;
        float x = 0.0f;
        if (k < NXX) {
            const float* pre; const float* K1;
            switch (term) {
                case 0:  pre = p0; K1 = k10; break;
                case 1:  pre = p1; K1 = k11; break;
                case 2:  pre = p2; K1 = k12; break;
                default: pre = p3; K1 = k13; break;
            }
            x = pre[k] * K1[k * NXX + j];
        }
        unsigned short hb, lb;
        split_bits(x, hb, lb);
        wsBth[r] = bc(hb);
        wsBtl[r] = bc(lb);
    }
}

// -------- Kernel 2: per-batch gather + split-bf16 WMMA GEMMs + weighted colsum --------
// P[b*80 + j] = sum_t colsum_i( A_t (x) (D_t @ B_t) )[j],  D = Rs (t<2) or Is (t>=2)
// D @ B  ~=  Dh@Bh + Dh@Bl + Dl@Bh   (f32 accumulate, ~1e-5 rel. err.)
__global__ __launch_bounds__(256) void eqnet_stage1(
    const float* __restrict__ inputs, const int* __restrict__ r_index,
    const float* __restrict__ wsA, const __bf16* __restrict__ wsBth,
    const __bf16* __restrict__ wsBtl, float* __restrict__ P)
{
    __shared__ __align__(16) __bf16 sRh[NXX * SSTR];
    __shared__ __align__(16) __bf16 sRl[NXX * SSTR];
    __shared__ __align__(16) __bf16 sIh[NXX * SSTR];
    __shared__ __align__(16) __bf16 sIl[NXX * SSTR];
    __shared__ float sOut[NXX];

    const int b   = blockIdx.x;
    const int tid = threadIdx.x;

    // gather + hi/lo split: Rs[i,k] = R[b, r_index[i*80+k]], Is likewise
    const float2* inp2 = (const float2*)inputs;
    for (int m = tid; m < NX2; m += 256) {
        int ri   = r_index[m];
        float2 v = inp2[(size_t)b * NX2 + ri];
        int row = m / NXX, col = m - row * NXX;
        int o = row * SSTR + col;
        unsigned short hb, lb;
        split_bits(v.x, hb, lb);  sRh[o] = bc(hb);  sRl[o] = bc(lb);
        split_bits(v.y, hb, lb);  sIh[o] = bc(hb);  sIl[o] = bc(lb);
    }
    // zero-pad K = 80..95
    const __bf16 z0 = bc((unsigned short)0);
    for (int zi = tid; zi < NXX * (KP - NXX); zi += 256) {
        int row = zi >> 4, col = NXX + (zi & 15);
        int o = row * SSTR + col;
        sRh[o] = z0; sRl[o] = z0; sIh[o] = z0; sIl[o] = z0;
    }
    if (tid < NXX) sOut[tid] = 0.0f;
    __syncthreads();

    const int wave = tid >> 5;
    const int lane = tid & 31;
    const int half = lane >> 4;
    const int l16  = lane & 15;

    // 4 terms x 25 tiles of 16x16; waves round-robin the 100 work items
    for (int t = wave; t < 100; t += 8) {
        const int term = t / 25;
        const int tile = t - term * 25;
        const int mi = tile / 5, ni = tile - mi * 5;
        const __bf16* Dh = (term < 2) ? sRh : sIh;
        const __bf16* Dl = (term < 2) ? sRl : sIl;
        const float* Amat = wsA + term * NX2;
        const int mrow = mi * 16 + l16;
        const int ncol = ni * 16 + l16;
        // B-operand: lane's column is ncol; K-run of 16 is contiguous in transposed B
        const __bf16* Bh = wsBth + ((size_t)term * NXX + ncol) * KP;
        const __bf16* Bl = wsBtl + ((size_t)term * NXX + ncol) * KP;

        v8f c = {0.f, 0.f, 0.f, 0.f, 0.f, 0.f, 0.f, 0.f};
        #pragma unroll
        for (int k0 = 0; k0 < KP; k0 += 32) {
            // A 16x32 bf16: lanes 0-15 -> K {0-7,16-23}; lanes 16-31 -> K {8-15,24-31}
            const int ao = mrow * SSTR + k0 + half * 8;
            v16bf ah = ld2x8(Dh + ao, 16);
            v16bf al = ld2x8(Dl + ao, 16);
            // B 32x16 bf16: lanes 0-15 -> K {0-7,16-23}; lanes 16-31 -> K {8-15,24-31}
            const int bo = k0 + half * 8;
            v16bf bh = ld2x8(Bh + bo, 16);
            v16bf bl = ld2x8(Bl + bo, 16);
            c = __builtin_amdgcn_wmma_f32_16x16x32_bf16(false, ah, false, bh, (short)0, c, false, false);
            c = __builtin_amdgcn_wmma_f32_16x16x32_bf16(false, ah, false, bl, (short)0, c, false, false);
            c = __builtin_amdgcn_wmma_f32_16x16x32_bf16(false, al, false, bh, (short)0, c, false, false);
        }
        // C layout: VGPR r, lane l -> M = r + 8*(l>=16), N = l&15
        float part = 0.f;
        #pragma unroll
        for (int r = 0; r < 8; ++r) {
            int i = mi * 16 + r + half * 8;
            part += c[r] * Amat[i * NXX + ncol];
        }
        atomicAdd(&sOut[ncol], part);   // ds_add_f32
    }
    __syncthreads();
    if (tid < NXX) P[(size_t)b * NXX + tid] = sOut[tid];
}

// -------- Kernel 3: out[g, n] = sum_m cart[n, m] * P[g, m]  (6400x6400 @ 6400x20) --------
// Memory-bound on cart_mat (164 MB) -> keep exact f32 WMMA, read cart exactly once.
__global__ __launch_bounds__(256) void eqnet_stage2(
    const float* __restrict__ cart, const float* __restrict__ P, float* __restrict__ out)
{
    __shared__ float red[8][2][256];

    const int n0   = blockIdx.x * 16;
    const int tid  = threadIdx.x;
    const int wave = tid >> 5;
    const int lane = tid & 31;
    const int half = lane >> 4;
    const int l16  = lane & 15;

    const int nrow = n0 + l16;
    const int g1   = 16 + l16;
    const float2* cart2 = (const float2*)(cart + (size_t)nrow * NX2);
    const float* prow0  = P + (size_t)l16 * NX2;
    const float* prow1  = (g1 < NG) ? (P + (size_t)g1 * NX2) : nullptr;

    v8f c0 = {0.f, 0.f, 0.f, 0.f, 0.f, 0.f, 0.f, 0.f};
    v8f c1 = {0.f, 0.f, 0.f, 0.f, 0.f, 0.f, 0.f, 0.f};

    const int kbeg = wave * (NX2 / 8);
    const int kend = kbeg + (NX2 / 8);
    #pragma unroll 4
    for (int k0 = kbeg; k0 < kend; k0 += 4) {
        const int ka = k0 + half * 2;          // even -> 8B aligned
        float2 av = cart2[ka >> 1];            // cart[nrow, ka..ka+1]
        v2f a;  a.x = av.x;  a.y = av.y;
        v2f b0; b0.x = prow0[ka]; b0.y = prow0[ka + 1];
        v2f b1;
        if (prow1) { b1.x = prow1[ka]; b1.y = prow1[ka + 1]; }
        else       { b1.x = 0.f;       b1.y = 0.f; }
        c0 = __builtin_amdgcn_wmma_f32_16x16x4_f32(false, a, false, b0, (short)0, c0, false, false);
        c1 = __builtin_amdgcn_wmma_f32_16x16x4_f32(false, a, false, b1, (short)0, c1, false, false);
    }

    #pragma unroll
    for (int r = 0; r < 8; ++r) {
        red[wave][0][r * 32 + lane] = c0[r];
        red[wave][1][r * 32 + lane] = c1[r];
    }
    __syncthreads();

    for (int e = tid; e < 512; e += 256) {
        int tl = e >> 8, pos = e & 255;
        float s = 0.f;
        #pragma unroll
        for (int w = 0; w < 8; ++w) s += red[w][tl][pos];
        int r = pos >> 5, l = pos & 31;
        int n = n0 + r + (l >> 4) * 8;
        int g = tl * 16 + (l & 15);
        if (g < NG) out[(size_t)g * NX2 + n] = s;
    }
}

extern "C" void kernel_launch(void* const* d_in, const int* in_sizes, int n_in,
                              void* d_out, int out_size, void* d_ws, size_t ws_size,
                              hipStream_t stream) {
    const float* inputs = (const float*)d_in[0];
    const float* cart   = (const float*)d_in[1];
    const float* pre1   = (const float*)d_in[2];
    const float* pre2   = (const float*)d_in[3];
    const float* pre3   = (const float*)d_in[4];
    const float* pre4   = (const float*)d_in[5];
    const float* post1  = (const float*)d_in[6];
    const float* post2  = (const float*)d_in[7];
    const float* post3  = (const float*)d_in[8];
    const float* post4  = (const float*)d_in[9];
    const float* cosk1  = (const float*)d_in[10];
    const float* sink1  = (const float*)d_in[11];
    const float* cosk2  = (const float*)d_in[12];
    const float* sink2  = (const float*)d_in[13];
    const float* cosk3  = (const float*)d_in[14];
    const float* sink3  = (const float*)d_in[15];
    const float* cosk4  = (const float*)d_in[16];
    const float* sink4  = (const float*)d_in[17];
    const int*   ridx   = (const int*)d_in[18];

    // ws layout (bytes):
    //   [0,          102400)  A planes, f32, 4*6400
    //   [102400,     163840)  Bt hi plane, bf16, 4*80*96
    //   [163840,     225280)  Bt lo plane, bf16, 4*80*96
    //   [225280,     737280)  P, f32, 128000 (viewed 20 x 6400)
    float*  wsA   = (float*)d_ws;
    __bf16* wsBth = (__bf16*)((char*)d_ws + 102400);
    __bf16* wsBtl = wsBth + NBT;
    float*  P     = (float*)((char*)d_ws + 225280);
    float*  out   = (float*)d_out;

    // term -> (post, K2=kernel2, pre, K1=kernel1):
    //  t0: post1, cos_kernel1, pre1, cos_kernel2   (data = Rs)
    //  t1: post2, sin_kernel1, pre2, sin_kernel2   (data = Rs)
    //  t2: post3, cos_kernel3, pre3, sin_kernel3   (data = Is)
    //  t3: post4, sin_kernel4, pre4, cos_kernel4   (data = Is)
    const int npre = 4 * NX2 + NBT;   // 25600 + 30720
    eqnet_precompute<<<(npre + 255) / 256, 256, 0, stream>>>(
        post1, post2, post3, post4,
        cosk1, sink1, cosk3, sink4,
        pre1, pre2, pre3, pre4,
        cosk2, sink2, sink3, cosk4,
        wsA, wsBth, wsBtl);

    eqnet_stage1<<<NBATCH, 256, 0, stream>>>(inputs, ridx, wsA, wsBth, wsBtl, P);

    eqnet_stage2<<<NX2 / 16, 256, 0, stream>>>(cart, P, out);
}
